// MyTransformerEncoderBlock_12326556139708
// MI455X (gfx1250) — matrix operand order repaired and checked by reading, not compiled, not run on hardware
//
#include <hip/hip_runtime.h>
#include <hip/hip_bf16.h>
#include <cstdint>

// ---------------------------------------------------------------------------
// Transformer encoder block for MI455X (gfx1250), bf16 WMMA everywhere,
// async global->LDS staging (ASYNCcnt) with double buffering in the GEMMs.
// B=4, S=2048, D=1024, H=16, dk=64, FF=4096. All GEMM dims divide tile sizes.
// ---------------------------------------------------------------------------

#define D_MODEL 1024
#define N_HEADS 16
#define DK      64
#define D_FF    4096
#define SEQ     2048
#define BATCH   4
#define ROWS    (BATCH * SEQ) /* 8192 */

typedef __attribute__((ext_vector_type(16))) __bf16   v16bf;
typedef __attribute__((ext_vector_type(8)))  float    v8f;
typedef __attribute__((ext_vector_type(8)))  unsigned v8u;

__device__ __forceinline__ unsigned short f32_to_bf16(float f) {
  unsigned u = __builtin_bit_cast(unsigned, f);
  u += 0x7FFFu + ((u >> 16) & 1u);       // round-to-nearest-even
  return (unsigned short)(u >> 16);
}

// D = A x B + C, A/B bf16 16x16x32, C/D f32 (8 VGPRs)
__device__ __forceinline__ v8f wmma_bf16(v16bf a, v16bf b, v8f c) {
  return __builtin_amdgcn_wmma_f32_16x16x32_bf16(false, a, false, b,
                                                 (short)0, c, false, false);
}

// ---------------------------------------------------------------------------
// CDNA5 async global -> LDS copy (per-lane 16B), tracked by ASYNCcnt.
// the LDS destination address comes from the vdst operand; low 32 bits of a generic pointer to
// __shared__ are the LDS byte offset (ISA 10.2 aperture mapping).
// ---------------------------------------------------------------------------
__device__ __forceinline__ void async_load_b128(unsigned lds_off,
                                                const void* gptr) {
  unsigned long long ga = (unsigned long long)(size_t)gptr;
  asm volatile("global_load_async_to_lds_b128 %0, %1, off"
               :: "v"(lds_off), "v"(ga)
               : "memory");
}
__device__ __forceinline__ void wait_asynccnt_le4() {
  asm volatile("s_wait_asynccnt 0x4" ::: "memory");
}
__device__ __forceinline__ void wait_asynccnt_0() {
  asm volatile("s_wait_asynccnt 0x0" ::: "memory");
}

// A fragment, 16x32 bf16 from row-major storage (ld elements, dword aligned).
// Layout: lane m = L&15; lanes 0-15 hold K=0..15, lanes 16-31 K=16..31.
__device__ __forceinline__ v16bf load_fragA(const unsigned short* p, int row,
                                            int ld, int k0) {
  const int lane = threadIdx.x & 31;
  const int kb = k0 + ((lane >> 4) << 4);
  const unsigned short* r = p + (size_t)row * ld;
  v8u u;
#pragma unroll
  for (int i = 0; i < 8; ++i) {
    int k = kb + 2 * i;
    u[i] = *(const unsigned*)(r + k);
  }
  return __builtin_bit_cast(v16bf, u);
}

// B fragment, 32x16 bf16 where consecutive K are contiguous in memory.
// Layout (same contiguous k order as load_fragA): lane n = L&15; lanes 0-15 hold K=0..15, lanes 16-31 K=16..31.
// rowp points at (k=0, n=this lane's column).
__device__ __forceinline__ v16bf load_fragB_contig(const unsigned short* rowp,
                                                   int k0) {
  const int lane = threadIdx.x & 31;
  const int kb = k0 + ((lane >> 4) << 4);
  v8u u;
#pragma unroll
  for (int i = 0; i < 8; ++i) u[i] = *(const unsigned*)(rowp + kb + 2 * i);
  return __builtin_bit_cast(v16bf, u);
}

// B fragment where the K dimension is strided (row-major K x N storage).
// colp points at (k=0) of this lane's column; element k at colp + k*strideK.
__device__ __forceinline__ v16bf load_fragB_strided(const unsigned short* colp,
                                                    int strideK) {
  const int lane = threadIdx.x & 31;
  const int kb = (lane >> 4) << 4;
  v8u u;
#pragma unroll
  for (int i = 0; i < 8; ++i) {
    unsigned lo = colp[(size_t)(kb + 2 * i) * strideK];
    unsigned hi = colp[(size_t)(kb + 2 * i + 1) * strideK];
    u[i] = lo | (hi << 16);
  }
  return __builtin_bit_cast(v16bf, u);
}

// ---------------------------------------------------------------------------
// f32 -> bf16 conversion (grid-stride)
// ---------------------------------------------------------------------------
__global__ void convert_f32_bf16_kernel(const float* __restrict__ in,
                                        unsigned short* __restrict__ out, int n) {
  int i = blockIdx.x * blockDim.x + threadIdx.x;
  int stride = gridDim.x * blockDim.x;
  for (; i < n; i += stride) out[i] = f32_to_bf16(in[i]);
}

// ---------------------------------------------------------------------------
// Tiled bf16 GEMM: C[M,N] = A[M,K] * B[K,N], row-major, M%128==N%128==K%32==0.
// 256 threads = 8 waves; waves arranged 2(M) x 4(N); each wave: 64x32 tile.
// Double-buffered async global->LDS staging: per stage each wave issues
// 4x global_load_async_to_lds_b128 (A: rows m0..m0+127 x 32k; B: 32k x 128n).
// ---------------------------------------------------------------------------
enum { EP_F32 = 0, EP_BF16 = 1, EP_BIAS_RELU_BF16 = 2, EP_BIAS_F32 = 3 };

template <int EP>
__global__ __launch_bounds__(256, 2) void gemm_bf16_kernel(
    const unsigned short* __restrict__ A, const unsigned short* __restrict__ B,
    float* __restrict__ Cf, unsigned short* __restrict__ Cb,
    const float* __restrict__ bias, int M, int N, int K) {
  __shared__ unsigned short sA[2][128 * 32];  // A tile, row-major [128m x 32k]
  __shared__ unsigned short sB[2][32 * 128];  // B tile, row-major [32k x 128n]

  const int t = threadIdx.x;
  const int lane = t & 31;
  const int wave = t >> 5;
  const int wm = wave >> 2;  // 0..1
  const int wn = wave & 3;   // 0..3
  const int m0 = blockIdx.y * 128;
  const int n0 = blockIdx.x * 128;

  // staging geometry (per thread, 16B chunks)
  const int ar = t >> 2;          // 0..63   A row within half-tile
  const int ac = (t & 3) * 8;     // 0,8,16,24
  const int br = t >> 4;          // 0..15   B row within half-tile
  const int bc = (t & 15) * 8;    // 0..120

  auto stage = [&](int buf, int k0) {
    async_load_b128((unsigned)(size_t)&sA[buf][ar * 32 + ac],
                    A + (size_t)(m0 + ar) * K + k0 + ac);
    async_load_b128((unsigned)(size_t)&sA[buf][(64 + ar) * 32 + ac],
                    A + (size_t)(m0 + 64 + ar) * K + k0 + ac);
    async_load_b128((unsigned)(size_t)&sB[buf][br * 128 + bc],
                    B + (size_t)(k0 + br) * N + n0 + bc);
    async_load_b128((unsigned)(size_t)&sB[buf][(16 + br) * 128 + bc],
                    B + (size_t)(k0 + 16 + br) * N + n0 + bc);
  };

  v8f acc[4][2];
#pragma unroll
  for (int i = 0; i < 4; ++i)
#pragma unroll
    for (int j = 0; j < 2; ++j) acc[i][j] = {};

  const int nk = K >> 5;
  stage(0, 0);
  for (int i = 0; i < nk; ++i) {
    const int cur = i & 1;
    const bool has_next = (i + 1) < nk;
    if (has_next) {
      stage(1 - cur, (i + 1) << 5);
      wait_asynccnt_le4();   // current stage's 4 ops complete (in-order)
    } else {
      wait_asynccnt_0();
    }
    __syncthreads();         // all waves' tile data visible

    v16bf af[4], bfrag[2];
#pragma unroll
    for (int ms = 0; ms < 4; ++ms)
      af[ms] = load_fragA(sA[cur], wm * 64 + ms * 16 + (lane & 15), 32, 0);
#pragma unroll
    for (int ns = 0; ns < 2; ++ns)
      bfrag[ns] =
          load_fragB_strided(sB[cur] + wn * 32 + ns * 16 + (lane & 15), 128);
#pragma unroll
    for (int ms = 0; ms < 4; ++ms)
#pragma unroll
      for (int ns = 0; ns < 2; ++ns)
        acc[ms][ns] = wmma_bf16(af[ms], bfrag[ns], acc[ms][ns]);
    __syncthreads();         // protect buf[cur] before next prefetch reuses it
  }

  // --- epilogue: C/D layout: VGPR i <-> m=(L>>4)*8+i, n=L&15 ---
  const int rbase = (lane >> 4) << 3;
  const int ncl = lane & 15;
#pragma unroll
  for (int ms = 0; ms < 4; ++ms) {
#pragma unroll
    for (int ns = 0; ns < 2; ++ns) {
      int gn = n0 + wn * 32 + ns * 16 + ncl;
      float bv = (EP == EP_BIAS_RELU_BF16 || EP == EP_BIAS_F32) ? bias[gn] : 0.0f;
#pragma unroll
      for (int i = 0; i < 8; ++i) {
        int gm = m0 + wm * 64 + ms * 16 + rbase + i;
        float v = acc[ms][ns][i] + bv;
        if (EP == EP_BIAS_RELU_BF16) v = v > 0.0f ? v : 0.0f;
        size_t off = (size_t)gm * N + gn;
        if (EP == EP_F32 || EP == EP_BIAS_F32)
          Cf[off] = v;
        else
          Cb[off] = f32_to_bf16(v);
      }
    }
  }
}

// ---------------------------------------------------------------------------
// Attention: one WG per (batch, head, 16-query tile). Two passes in LDS:
//   pass1: S[16,2048] = (Q K^T) * 1/sqrt(dk)   (WMMA, f32 in LDS)
//   softmax in LDS -> P bf16
//   pass2: ctx[16,64] = P V                    (WMMA)
// Dynamic LDS: 16*2048*4 + 16*2048*2 + 16*64*2 + 16*4 = 198720 B (< 320 KB)
// ---------------------------------------------------------------------------
__global__ __launch_bounds__(256, 1) void attention_kernel(
    const unsigned short* __restrict__ Q, const unsigned short* __restrict__ K,
    const unsigned short* __restrict__ V, unsigned short* __restrict__ CTX) {
  extern __shared__ char smem[];
  float* sS = (float*)smem;                                       // 16 x 2048 f32
  unsigned short* sP = (unsigned short*)(smem + 16 * SEQ * 4);    // 16 x 2048 bf16
  unsigned short* sQ = (unsigned short*)(smem + 16 * SEQ * 6);    // 16 x 64 bf16
  float* sSum = (float*)(smem + 16 * SEQ * 6 + 16 * DK * 2);      // 16 f32

  const int qt = blockIdx.x;  // query tile 0..127
  const int h = blockIdx.y;   // head
  const int b = blockIdx.z;   // batch
  const int q0 = qt * 16;
  const int t = threadIdx.x;
  const int lane = t & 31;
  const int wave = t >> 5;

  // stage Q tile (16 x 64 bf16)
  for (int e = t * 2; e < 16 * DK; e += 512) {
    int r = e >> 6, c = e & 63;
    *(unsigned*)(&sQ[e]) =
        *(const unsigned*)(Q + (size_t)(b * SEQ + q0 + r) * D_MODEL + h * DK + c);
  }
  __syncthreads();

  v16bf qf0 = load_fragA(sQ, lane & 15, DK, 0);
  v16bf qf1 = load_fragA(sQ, lane & 15, DK, 32);

  // ---- pass 1: scores; wave w covers key columns [w*256, w*256+256) ----
  const float scale = 0.125f;  // 1/sqrt(64)
  const int key_n = lane & 15;
  const int rbase = (lane >> 4) << 3;
  for (int nt = 0; nt < 16; ++nt) {
    int ncol0 = wave * 256 + nt * 16;
    // B[kd][n] = K[key=ncol0+n][kd]; kd contiguous in K's row-major storage
    const unsigned short* krow =
        K + (size_t)(b * SEQ + ncol0 + key_n) * D_MODEL + h * DK;
    v16bf bf0 = load_fragB_contig(krow, 0);
    v16bf bf1 = load_fragB_contig(krow, 32);
    v8f acc = {};
    acc = wmma_bf16(qf0, bf0, acc);
    acc = wmma_bf16(qf1, bf1, acc);
#pragma unroll
    for (int i = 0; i < 8; ++i)
      sS[(size_t)(rbase + i) * SEQ + ncol0 + key_n] = acc[i] * scale;
  }
  __syncthreads();

  // ---- softmax: 16 threads per row, row r = t>>4 ----
  {
    int r = t >> 4;
    int c0 = t & 15;
    float mx = -3.0e38f;
    for (int c = c0; c < SEQ; c += 16) mx = fmaxf(mx, sS[(size_t)r * SEQ + c]);
#pragma unroll
    for (int off = 1; off < 16; off <<= 1) mx = fmaxf(mx, __shfl_xor(mx, off, 32));
    float sum = 0.0f;
    for (int c = c0; c < SEQ; c += 16) {
      float e = __expf(sS[(size_t)r * SEQ + c] - mx);
      sum += e;
      sP[(size_t)r * SEQ + c] = f32_to_bf16(e);
    }
#pragma unroll
    for (int off = 1; off < 16; off <<= 1) sum += __shfl_xor(sum, off, 32);
    if (c0 == 0) sSum[r] = sum;
  }
  __syncthreads();

  // ---- pass 2: ctx = P @ V (16 x 64); waves 0..3, one 16-col tile each ----
  if (wave < 4) {
    const int n = lane & 15;
    const unsigned short* vcol =
        V + (size_t)(b * SEQ) * D_MODEL + h * DK + wave * 16 + n;
    v8f acc = {};
    for (int kt = 0; kt < SEQ / 32; ++kt) {
      v16bf a = load_fragA(sP, lane & 15, SEQ, kt * 32);
      v16bf bb = load_fragB_strided(vcol + (size_t)kt * 32 * D_MODEL, D_MODEL);
      acc = wmma_bf16(a, bb, acc);
    }
#pragma unroll
    for (int i = 0; i < 8; ++i) {
      float v = acc[i] / sSum[rbase + i];
      CTX[(size_t)(b * SEQ + q0 + rbase + i) * D_MODEL + h * DK + wave * 16 + n] =
          f32_to_bf16(v);
    }
  }
}

// ---------------------------------------------------------------------------
// Fused residual add + LayerNorm (population variance, eps=1e-5).
// One block per row of 1024; writes f32 and optionally bf16.
// ---------------------------------------------------------------------------
__global__ __launch_bounds__(256) void add_ln_kernel(
    const float* __restrict__ X, const float* __restrict__ Y,
    const float* __restrict__ gamma, const float* __restrict__ beta,
    float* __restrict__ outf, unsigned short* __restrict__ outb) {
  const int row = blockIdx.x;
  const int t = threadIdx.x;
  float vals[4];
  float s = 0.0f, s2 = 0.0f;
#pragma unroll
  for (int j = 0; j < 4; ++j) {
    int c = t + j * 256;
    float v = X[(size_t)row * D_MODEL + c] + Y[(size_t)row * D_MODEL + c];
    vals[j] = v;
    s += v;
    s2 += v * v;
  }
#pragma unroll
  for (int off = 16; off >= 1; off >>= 1) {
    s += __shfl_xor(s, off, 32);
    s2 += __shfl_xor(s2, off, 32);
  }
  __shared__ float ls[8], ls2[8];
  __shared__ float mu_s, rstd_s;
  if ((t & 31) == 0) {
    ls[t >> 5] = s;
    ls2[t >> 5] = s2;
  }
  __syncthreads();
  if (t == 0) {
    float a = 0.0f, bsum = 0.0f;
#pragma unroll
    for (int i = 0; i < 8; ++i) {
      a += ls[i];
      bsum += ls2[i];
    }
    float mu = a * (1.0f / D_MODEL);
    float var = bsum * (1.0f / D_MODEL) - mu * mu;
    mu_s = mu;
    rstd_s = rsqrtf(var + 1e-5f);
  }
  __syncthreads();
  const float mu = mu_s, rstd = rstd_s;
#pragma unroll
  for (int j = 0; j < 4; ++j) {
    int c = t + j * 256;
    float o = (vals[j] - mu) * rstd * gamma[c] + beta[c];
    outf[(size_t)row * D_MODEL + c] = o;
    if (outb) outb[(size_t)row * D_MODEL + c] = f32_to_bf16(o);
  }
}

// ---------------------------------------------------------------------------
// Launch
// ---------------------------------------------------------------------------
extern "C" void kernel_launch(void* const* d_in, const int* in_sizes, int n_in,
                              void* d_out, int out_size, void* d_ws,
                              size_t ws_size, hipStream_t stream) {
  const float* x = (const float*)d_in[0];
  const float* wq = (const float*)d_in[1];
  const float* wk = (const float*)d_in[2];
  const float* wv = (const float*)d_in[3];
  const float* wo = (const float*)d_in[4];
  const float* w1 = (const float*)d_in[5];
  const float* b1 = (const float*)d_in[6];
  const float* w2 = (const float*)d_in[7];
  const float* b2 = (const float*)d_in[8];
  const float* g1 = (const float*)d_in[9];
  const float* be1 = (const float*)d_in[10];
  const float* g2 = (const float*)d_in[11];
  const float* be2 = (const float*)d_in[12];
  float* out = (float*)d_out;
  (void)in_sizes; (void)n_in; (void)out_size; (void)ws_size;

  char* ws = (char*)d_ws;
  const size_t MB = 1024ull * 1024ull;
  unsigned short* xb   = (unsigned short*)(ws + 0);        // 16 MB
  unsigned short* wqb  = (unsigned short*)(ws + 16 * MB);  // 2 MB
  unsigned short* wkb  = (unsigned short*)(ws + 18 * MB);
  unsigned short* wvb  = (unsigned short*)(ws + 20 * MB);
  unsigned short* wob  = (unsigned short*)(ws + 22 * MB);
  unsigned short* w1b  = (unsigned short*)(ws + 24 * MB);  // 8 MB
  unsigned short* w2b  = (unsigned short*)(ws + 32 * MB);  // 8 MB
  unsigned short* qb   = (unsigned short*)(ws + 40 * MB);  // 16 MB
  unsigned short* kbuf = (unsigned short*)(ws + 56 * MB);  // 16 MB
  unsigned short* vbuf = (unsigned short*)(ws + 72 * MB);  // 16 MB
  unsigned short* ctxb = (unsigned short*)(ws + 88 * MB);  // 16 MB
  float* attnf         = (float*)(ws + 104 * MB);          // 32 MB
  float* hf            = (float*)(ws + 136 * MB);          // 32 MB
  unsigned short* hb   = (unsigned short*)(ws + 168 * MB); // 16 MB
  unsigned short* a1b  = (unsigned short*)(ws + 184 * MB); // 64 MB
  float* f2f           = (float*)(ws + 248 * MB);          // 32 MB

  // --- bf16 conversions ---
  auto cvt = [&](const float* src, unsigned short* dst, int n) {
    convert_f32_bf16_kernel<<<dim3(1024), dim3(256), 0, stream>>>(src, dst, n);
  };
  cvt(x, xb, ROWS * D_MODEL);
  cvt(wq, wqb, D_MODEL * D_MODEL);
  cvt(wk, wkb, D_MODEL * D_MODEL);
  cvt(wv, wvb, D_MODEL * D_MODEL);
  cvt(wo, wob, D_MODEL * D_MODEL);
  cvt(w1, w1b, D_MODEL * D_FF);
  cvt(w2, w2b, D_FF * D_MODEL);

  // --- QKV projections ---
  dim3 gproj(D_MODEL / 128, ROWS / 128);
  gemm_bf16_kernel<EP_BF16><<<gproj, 256, 0, stream>>>(
      xb, wqb, nullptr, qb, nullptr, ROWS, D_MODEL, D_MODEL);
  gemm_bf16_kernel<EP_BF16><<<gproj, 256, 0, stream>>>(
      xb, wkb, nullptr, kbuf, nullptr, ROWS, D_MODEL, D_MODEL);
  gemm_bf16_kernel<EP_BF16><<<gproj, 256, 0, stream>>>(
      xb, wvb, nullptr, vbuf, nullptr, ROWS, D_MODEL, D_MODEL);

  // --- attention ---
  const size_t attn_lds =
      (size_t)16 * SEQ * 4 + (size_t)16 * SEQ * 2 + (size_t)16 * DK * 2 + 16 * 4;
  hipFuncSetAttribute((const void*)attention_kernel,
                      hipFuncAttributeMaxDynamicSharedMemorySize, (int)attn_lds);
  attention_kernel<<<dim3(SEQ / 16, N_HEADS, BATCH), 256, attn_lds, stream>>>(
      qb, kbuf, vbuf, ctxb);

  // --- output projection ---
  gemm_bf16_kernel<EP_F32><<<gproj, 256, 0, stream>>>(
      ctxb, wob, attnf, nullptr, nullptr, ROWS, D_MODEL, D_MODEL);

  // --- add & norm 1 ---
  add_ln_kernel<<<dim3(ROWS), 256, 0, stream>>>(x, attnf, g1, be1, hf, hb);

  // --- FFN ---
  gemm_bf16_kernel<EP_BIAS_RELU_BF16>
      <<<dim3(D_FF / 128, ROWS / 128), 256, 0, stream>>>(hb, w1b, nullptr, a1b,
                                                         b1, ROWS, D_FF, D_MODEL);
  gemm_bf16_kernel<EP_BIAS_F32><<<gproj, 256, 0, stream>>>(
      a1b, w2b, f2f, nullptr, b2, ROWS, D_MODEL, D_FF);

  // --- add & norm 2 ---
  add_ln_kernel<<<dim3(ROWS), 256, 0, stream>>>(hf, f2f, g2, be2, out, nullptr);
}
